// SelfAttention1d_4578435138036
// MI455X (gfx1250) — matrix operand order repaired and checked by reading, not compiled, not run on hardware
//
#include <hip/hip_runtime.h>
#include <hip/hip_bf16.h>

typedef _Float16 half_t;
typedef __attribute__((ext_vector_type(16))) _Float16 v16h;
typedef __attribute__((ext_vector_type(8)))  _Float16 v8h;
typedef __attribute__((ext_vector_type(8)))  float    v8f;

#define CC 256
#define HH 8
#define BB 2
#define SS 4096
#define DD 32
#define EPSF 1e-5f

// ---------------- WMMA fragment helpers (ISA 7.12.2 layouts) ----------------

// A-matrix 16x32 f16: lane (l&15)=row M; lanes<16 hold K={0..7,16..23}, lanes>=16 K={8..15,24..31}
static __device__ __forceinline__ v16h load_frag_a(const half_t* __restrict__ p, int ld,
                                                   int row, int kb, int lane) {
  const int grp = lane >> 4;
  const half_t* base = p + (size_t)(row + (lane & 15)) * ld + kb + grp * 8;
  v8h lo = *(const v8h*)(base);
  v8h hi = *(const v8h*)(base + 16);
  v16h f;
#pragma unroll
  for (int i = 0; i < 8; ++i) { f[i] = lo[i]; f[i + 8] = hi[i]; }
  return f;
}

// B-matrix 32x16 f16 from n-major storage (row n holds all K contiguously):
// lane (l&15)=col N; lanes<16 hold K={0..7,16..23}, lanes>=16 K={8..15,24..31}
static __device__ __forceinline__ v16h load_frag_b(const half_t* __restrict__ p, int ld,
                                                   int n0, int kb, int lane) {
  const int grp = lane >> 4;
  const half_t* base = p + (size_t)(n0 + (lane & 15)) * ld + kb + grp * 8;
  v8h lo = *(const v8h*)(base);
  v8h hi = *(const v8h*)(base + 16);
  v16h f;
#pragma unroll
  for (int i = 0; i < 8; ++i) { f[i] = lo[i]; f[i + 8] = hi[i]; }
  return f;
}

static __device__ __forceinline__ v8f wmma_f16(v16h a, v16h b, v8f c) {
  return __builtin_amdgcn_wmma_f32_16x16x32_f16(false, a, false, b, (short)0, c, false, false);
}

// reductions across one 16-lane group (row of the C/D layout); xor masks <=8 stay in-half
static __device__ __forceinline__ float grp16_max(float v) {
  v = fmaxf(v, __shfl_xor(v, 1, 32));
  v = fmaxf(v, __shfl_xor(v, 2, 32));
  v = fmaxf(v, __shfl_xor(v, 4, 32));
  v = fmaxf(v, __shfl_xor(v, 8, 32));
  return v;
}
static __device__ __forceinline__ float grp16_sum(float v) {
  v += __shfl_xor(v, 1, 32);
  v += __shfl_xor(v, 2, 32);
  v += __shfl_xor(v, 4, 32);
  v += __shfl_xor(v, 8, 32);
  return v;
}

// ---------------- Kernel 1: GroupNorm(1 group) statistics ----------------
__global__ void gn_stats_kernel(const float* __restrict__ x, float* __restrict__ stats) {
  const int b = blockIdx.x >> 7;     // 128 blocks per batch element
  const int blk = blockIdx.x & 127;
  const float* xb = x + (size_t)b * CC * SS;
  float s = 0.f, s2 = 0.f;
  for (int i = blk * 256 + threadIdx.x; i < CC * SS; i += 128 * 256) {
    float v = xb[i]; s += v; s2 += v * v;
  }
#pragma unroll
  for (int off = 16; off; off >>= 1) { s += __shfl_xor(s, off, 32); s2 += __shfl_xor(s2, off, 32); }
  __shared__ float red[2][8];
  const int wave = threadIdx.x >> 5;
  if ((threadIdx.x & 31) == 0) { red[0][wave] = s; red[1][wave] = s2; }
  __syncthreads();
  if (threadIdx.x == 0) {
    float ts = 0.f, t2 = 0.f;
#pragma unroll
    for (int w = 0; w < 8; ++w) { ts += red[0][w]; t2 += red[1][w]; }
    atomicAdd(&stats[b * 2 + 0], ts);
    atomicAdd(&stats[b * 2 + 1], t2);
  }
}

// ---------------- Kernel 2: normalize + transpose to [B,S,C] f16 ----------------
__global__ void gn_norm_kernel(const float* __restrict__ x, const float* __restrict__ w,
                               const float* __restrict__ bias, const float* __restrict__ stats,
                               half_t* __restrict__ h16) {
  const size_t idx = (size_t)blockIdx.x * 256 + threadIdx.x;   // over B*C*S, s fastest
  const int b = (int)(idx / ((size_t)CC * SS));
  const int rem = (int)(idx % ((size_t)CC * SS));
  const int c = rem / SS;
  const int s = rem % SS;
  const float invn = 1.f / (float)(CC * SS);
  const float mean = stats[b * 2 + 0] * invn;
  const float var = stats[b * 2 + 1] * invn - mean * mean;
  const float rinv = rsqrtf(var + EPSF);
  const float v = (x[idx] - mean) * rinv * w[c] + bias[c];
  h16[((size_t)b * SS + s) * CC + c] = (half_t)v;
}

// ---------------- Kernel 0: weights f32 -> f16 ----------------
__global__ void wcvt_kernel(const float* p0, const float* p1, const float* p2, const float* p3,
                            half_t* d0, half_t* d1, half_t* d2, half_t* d3) {
  const int i = blockIdx.x * 256 + threadIdx.x;   // 0..C*C-1
  const float* src = (blockIdx.y == 0) ? p0 : (blockIdx.y == 1) ? p1 : (blockIdx.y == 2) ? p2 : p3;
  half_t* dst = (blockIdx.y == 0) ? d0 : (blockIdx.y == 1) ? d1 : (blockIdx.y == 2) ? d2 : d3;
  dst[i] = (half_t)src[i];
}

// ---------------- Kernel 3/5: WMMA GEMM  out = A[M,K] * W[N,K]^T + bias ----------------
// block = 8 waves; each wave: 16 (M) x 64 (N); block tile 128x64; K loop step 32.
// mode 0: f16 out [m][C] * scale   mode 1: f16 out transposed [b][n][s]
// mode 2: f32 out transposed [b][n][s] + residual
__global__ void gemm_kernel(const half_t* __restrict__ A, const half_t* __restrict__ W16,
                            const float* __restrict__ bias,
                            half_t* __restrict__ out16, float* __restrict__ out32,
                            const float* __restrict__ resid, float scale, int mode) {
  const int lane = threadIdx.x & 31;
  const int wave = threadIdx.x >> 5;
  const int grp = lane >> 4;
  const int m0 = blockIdx.y * 128 + wave * 16;
  const int n0 = blockIdx.x * 64;
  v8f acc[4] = {};
  for (int kb = 0; kb < CC; kb += 32) {
    v16h a = load_frag_a(A, CC, m0, kb, lane);
#pragma unroll
    for (int t = 0; t < 4; ++t) {
      v16h bm = load_frag_b(W16, CC, n0 + t * 16, kb, lane);
      acc[t] = wmma_f16(a, bm, acc[t]);
    }
  }
#pragma unroll
  for (int t = 0; t < 4; ++t) {
    const int n = n0 + t * 16 + (lane & 15);
    const float bv = bias[n];
#pragma unroll
    for (int r = 0; r < 8; ++r) {
      const int m = m0 + r + 8 * grp;          // C/D layout: row = r + 8*(lane>>4)
      const float v = acc[t][r] + bv;
      if (mode == 0) {
        out16[(size_t)m * CC + n] = (half_t)(v * scale);
      } else if (mode == 1) {
        const int bb = m >> 12, s = m & (SS - 1);
        out16[((size_t)bb * CC + n) * SS + s] = (half_t)v;
      } else {
        const int bb = m >> 12, s = m & (SS - 1);
        const size_t o = ((size_t)bb * CC + n) * SS + s;
        out32[o] = v + resid[o];
      }
    }
  }
}

// ---------------- Kernel 4: flash attention ----------------
// grid (S/128, H, B); block = 8 waves, 16 query rows per wave; key chunks of 32.
__global__ void attn_kernel(const half_t* __restrict__ q16, const half_t* __restrict__ k16,
                            const half_t* __restrict__ vT16, half_t* __restrict__ o16) {
  __shared__ half_t ldsP[8][16 * 32];   // per-wave P tile staging (8 KB)
  const int lane = threadIdx.x & 31;
  const int wave = threadIdx.x >> 5;
  const int grp = lane >> 4;
  const int b = blockIdx.z, h = blockIdx.y;
  const int q0 = blockIdx.x * 128 + wave * 16;
  const half_t* qb = q16 + (size_t)b * SS * CC;
  const half_t* kp = k16 + (size_t)b * SS * CC;
  const half_t* vTb = vT16 + (size_t)b * CC * SS;
  half_t* ob = o16 + (size_t)b * SS * CC;
  half_t* myP = &ldsP[wave][0];

  const v16h aq = load_frag_a(qb, CC, q0, h * DD, lane);   // Q: 16x32, full d_head in K
  v8f o0 = {}; v8f o1 = {};
  float mrow[8], lrow[8];
#pragma unroll
  for (int r = 0; r < 8; ++r) { mrow[r] = -3.0e38f; lrow[r] = 0.f; }

  for (int k0 = 0; k0 < SS; k0 += 32) {
    v16h bk0 = load_frag_b(kp, CC, k0, h * DD, lane);        // K^T: B(d, key)
    v16h bk1 = load_frag_b(kp, CC, k0 + 16, h * DD, lane);
    v8f z = {};
    v8f s0 = wmma_f16(aq, bk0, z);
    v8f s1 = wmma_f16(aq, bk1, z);
    if (k0 + 32 < SS)
      __builtin_prefetch(kp + (size_t)(k0 + 32 + (lane & 15)) * CC + h * DD, 0, 1);

#pragma unroll
    for (int r = 0; r < 8; ++r) {
      float mx = grp16_max(fmaxf(s0[r], s1[r]));
      const float mnew = fmaxf(mrow[r], mx);
      const float corr = __expf(mrow[r] - mnew);
      const float p0 = __expf(s0[r] - mnew);
      const float p1 = __expf(s1[r] - mnew);
      const float ps = grp16_sum(p0 + p1);
      lrow[r] = lrow[r] * corr + ps;
      mrow[r] = mnew;
      o0[r] *= corr;
      o1[r] *= corr;
      const int m = r + 8 * grp;
      myP[m * 32 + (lane & 15)] = (half_t)p0;        // P chunk 16x32 in LDS, row-major
      myP[m * 32 + 16 + (lane & 15)] = (half_t)p1;
    }
#if __has_builtin(__builtin_amdgcn_s_wait_dscnt)
    __builtin_amdgcn_s_wait_dscnt(0);
#else
    asm volatile("s_wait_dscnt 0" ::: "memory");
#endif
    const v16h ap = load_frag_a(myP, 32, 0, 0, lane);        // P as A-fragment (cross-lane via LDS)
    const v16h bv0 = load_frag_b(vTb, SS, h * DD, k0, lane); // V^T rows are d, contiguous in keys
    const v16h bv1 = load_frag_b(vTb, SS, h * DD + 16, k0, lane);
    o0 = wmma_f16(ap, bv0, o0);
    o1 = wmma_f16(ap, bv1, o1);
  }

#pragma unroll
  for (int r = 0; r < 8; ++r) {
    const float inv = 1.f / lrow[r];
    const int s = q0 + r + 8 * grp;
    ob[(size_t)s * CC + h * DD + (lane & 15)] = (half_t)(o0[r] * inv);
    ob[(size_t)s * CC + h * DD + 16 + (lane & 15)] = (half_t)(o1[r] * inv);
  }
}

// ---------------- launch ----------------
extern "C" void kernel_launch(void* const* d_in, const int* in_sizes, int n_in,
                              void* d_out, int out_size, void* d_ws, size_t ws_size,
                              hipStream_t stream) {
  (void)in_sizes; (void)n_in; (void)out_size; (void)ws_size;
  const float* hidden = (const float*)d_in[0];
  const float* gnw = (const float*)d_in[1];
  const float* gnb = (const float*)d_in[2];
  const float* Wq = (const float*)d_in[3];
  const float* bq = (const float*)d_in[4];
  const float* Wk = (const float*)d_in[5];
  const float* bk = (const float*)d_in[6];
  const float* Wv = (const float*)d_in[7];
  const float* bv = (const float*)d_in[8];
  const float* Wo = (const float*)d_in[9];
  const float* bo = (const float*)d_in[10];
  float* out = (float*)d_out;

  char* w = (char*)d_ws;
  float* stats = (float*)w;
  size_t off = 256;
  half_t* Wq16 = (half_t*)(w + off); off += (size_t)CC * CC * 2;
  half_t* Wk16 = (half_t*)(w + off); off += (size_t)CC * CC * 2;
  half_t* Wv16 = (half_t*)(w + off); off += (size_t)CC * CC * 2;
  half_t* Wo16 = (half_t*)(w + off); off += (size_t)CC * CC * 2;
  const size_t TEN = (size_t)BB * SS * CC * 2;
  half_t* h16 = (half_t*)(w + off); off += TEN;
  half_t* q16 = (half_t*)(w + off); off += TEN;
  half_t* k16 = (half_t*)(w + off); off += TEN;
  half_t* vT16 = (half_t*)(w + off); off += TEN;
  half_t* attn16 = (half_t*)(w + off); off += TEN;

  hipMemsetAsync(stats, 0, 16 * sizeof(float), stream);
  wcvt_kernel<<<dim3(CC * CC / 256, 4), 256, 0, stream>>>(Wq, Wk, Wv, Wo, Wq16, Wk16, Wv16, Wo16);
  gn_stats_kernel<<<BB * 128, 256, 0, stream>>>(hidden, stats);
  gn_norm_kernel<<<(BB * CC * SS) / 256, 256, 0, stream>>>(hidden, gnw, gnb, stats, h16);

  const float scale = 0.42044820762685725f;   // 32^(-1/4), folded into q and k
  gemm_kernel<<<dim3(CC / 64, (BB * SS) / 128), 256, 0, stream>>>(h16, Wq16, bq, q16, nullptr, nullptr, scale, 0);
  gemm_kernel<<<dim3(CC / 64, (BB * SS) / 128), 256, 0, stream>>>(h16, Wk16, bk, k16, nullptr, nullptr, scale, 0);
  gemm_kernel<<<dim3(CC / 64, (BB * SS) / 128), 256, 0, stream>>>(h16, Wv16, bv, vT16, nullptr, nullptr, 1.0f, 1);

  attn_kernel<<<dim3(SS / 128, HH, BB), 256, 0, stream>>>(q16, k16, vT16, attn16);

  gemm_kernel<<<dim3(CC / 64, (BB * SS) / 128), 256, 0, stream>>>(attn16, Wo16, bo, nullptr, out, hidden, 1.0f, 2);
}
